// SASRec_69114613729798
// MI455X (gfx1250) — matrix operand order repaired and checked by reading, not compiled, not run on hardware
//
#include <hip/hip_runtime.h>
#include <hip/hip_bf16.h>

// SASRec forward for MI455X (gfx1250, wave32).
// GEMMs: v_wmma_f32_16x16x32_f16, 16x64 tile per wave, weight tile staged in
// LDS via global_load_async_to_lds_b128 (ASYNCcnt) when available.
// Attention / LN / k-means are wave32 VALU+LDS kernels.

#define BB 128
#define SS 200
#define DD 256
#define HH 4
#define DHD 64
#define LLAYERS 2
#define KKC 4
#define KM_ITERS 5
#define EPSLN 1e-8f

typedef _Float16 f16;
typedef __attribute__((ext_vector_type(16))) _Float16 v16h;
typedef __attribute__((ext_vector_type(8)))  float    v8f;
typedef int v4i_g __attribute__((vector_size(16)));   // matches builtin param type

#ifndef __has_builtin
#define __has_builtin(x) 0
#endif
#if __has_builtin(__builtin_amdgcn_global_load_async_to_lds_b128) && \
    __has_builtin(__builtin_amdgcn_s_wait_asynccnt)
#define USE_ASYNC_LDS 1
#else
#define USE_ASYNC_LDS 0
#endif

// ---------------------------------------------------------------- embedding
__global__ void embed_kernel(const int* __restrict__ logs,
                             const float* __restrict__ item_emb,
                             const float* __restrict__ pos_emb,
                             float* __restrict__ seqs) {
  int idx = blockIdx.x * blockDim.x + threadIdx.x;
  const int total = BB * SS * DD;
  if (idx >= total) return;
  int d  = idx & (DD - 1);
  int bs = idx >> 8;            // idx / DD
  int s  = bs % SS;
  int item = logs[bs];
  float v = 0.0f;
  if (item != 0)                // mask folds pos_emb to zero too
    v = item_emb[(size_t)item * DD + d] * 16.0f + pos_emb[s * DD + d];
  seqs[idx] = v;
}

// ---------------------------------------------------------------- layernorm
// one wave per row of 256; 8 waves per block
__global__ __launch_bounds__(256)
void ln_kernel(const float* __restrict__ x, const float* __restrict__ g,
               const float* __restrict__ b, float* __restrict__ out, int rows) {
  const int wave = threadIdx.x >> 5, lane = threadIdx.x & 31;
  const int r = blockIdx.x * 8 + wave;
  if (r >= rows) return;
  const float* xr = x + (size_t)r * DD;
  float v[8];
  float sum = 0.0f;
#pragma unroll
  for (int j = 0; j < 8; ++j) { v[j] = xr[lane + 32 * j]; sum += v[j]; }
#pragma unroll
  for (int off = 16; off > 0; off >>= 1) sum += __shfl_xor(sum, off, 32);
  const float mu = sum * (1.0f / DD);
  float vs = 0.0f;
#pragma unroll
  for (int j = 0; j < 8; ++j) { float t = v[j] - mu; vs += t * t; }
#pragma unroll
  for (int off = 16; off > 0; off >>= 1) vs += __shfl_xor(vs, off, 32);
  const float rs = rsqrtf(vs * (1.0f / DD) + EPSLN);
  float* orow = out + (size_t)r * DD;
#pragma unroll
  for (int j = 0; j < 8; ++j) {
    int d = lane + 32 * j;
    orow[d] = (v[j] - mu) * rs * g[d] + b[d];
  }
}

// ---------------------------------------------------------------- f32 -> f16
__global__ void cvt_f32_to_f16(const float* __restrict__ in,
                               f16* __restrict__ out, int n) {
  int i = blockIdx.x * blockDim.x + threadIdx.x;
  if (i < n) out[i] = (f16)in[i];
}

// ---------------------------------------------------------------- WMMA GEMM
// C[M,N] = A[M,K] @ W[N,K]^T + bias, with fused epilogue.
// MODE 0: bias   1: bias+relu   2: bias+residual   3: bias+residual+seq-mask
// Block = 8 waves stacked along M; block tile 128x64.
// Wave computes 16x64 (4 n-tiles): A fragment reused across 4 WMMAs.
// W tile (64 rows x 256 k, f16, 32 KB) staged in LDS, shared by all 8 waves.
template <int MODE>
__global__ __launch_bounds__(256)
void gemm_wmma_f16(const f16* __restrict__ A, const f16* __restrict__ W,
                   const float* __restrict__ bias, const float* __restrict__ Res,
                   const int* __restrict__ logs, float* __restrict__ Cout,
                   int M, int N, int Kd) {
  __shared__ f16 sW[64][DD];                       // 32 KB
  const int tid  = threadIdx.x;
  const int lane = tid & 31;
  const int wave = tid >> 5;
  const int m0 = blockIdx.x * 128 + wave * 16;
  const int nb = blockIdx.y * 64;

  // ---- stage weight tile rows nb..nb+63 (all K) into LDS ----
  {
    const f16* wg = W + (size_t)nb * Kd;
#if USE_ASYNC_LDS
#pragma unroll
    for (int j = 0; j < 8; ++j) {
      const int c   = tid + 256 * j;               // 16-byte chunk id
      const int row = c >> 5;                      // 32 chunks per row
      const int ko  = (c & 31) << 3;               // 8 halves per chunk
      __builtin_amdgcn_global_load_async_to_lds_b128(
          (__attribute__((address_space(1))) v4i_g*)(wg + (size_t)row * Kd + ko),
          (__attribute__((address_space(3))) v4i_g*)&sW[row][ko],
          /*offset=*/0, /*cpol=*/0);
    }
    __builtin_amdgcn_s_wait_asynccnt(0);
#else
#pragma unroll
    for (int j = 0; j < 8; ++j) {
      const int c   = tid + 256 * j;
      const int row = c >> 5;
      const int ko  = (c & 31) << 3;
      *(int4*)&sW[row][ko] = *(const int4*)(wg + (size_t)row * Kd + ko);
    }
#endif
    __syncthreads();
  }

  const int half = lane >> 4;                      // 0: lanes 0-15, 1: 16-31
  const int l15  = lane & 15;
  // A fragment (16x32 f16): lane holds row m0+l15, k = half*16+{0..15}
  const f16* arow = A + (size_t)(m0 + l15) * Kd;
  v8f acc0 = {}, acc1 = {}, acc2 = {}, acc3 = {};
  for (int k0 = 0; k0 < Kd; k0 += 32) {
    v16h af;
    const f16* ap = arow + k0 + half * 16;
#pragma unroll
    for (int j = 0; j < 8; ++j) { af[j] = ap[j]; af[j + 8] = ap[8 + j]; }
    // B fragments from LDS: B[k][n] = W[n][k]; lane holds 16 consecutive k of row n
    const int kb = k0 + half * 16;
    v16h b0, b1, b2, b3;
#pragma unroll
    for (int j = 0; j < 16; ++j) {
      b0[j] = sW[l15     ][kb + j];
      b1[j] = sW[l15 + 16][kb + j];
      b2[j] = sW[l15 + 32][kb + j];
      b3[j] = sW[l15 + 48][kb + j];
    }
    acc0 = __builtin_amdgcn_wmma_f32_16x16x32_f16(false, af, false, b0, (short)0, acc0, false, false);
    acc1 = __builtin_amdgcn_wmma_f32_16x16x32_f16(false, af, false, b1, (short)0, acc1, false, false);
    acc2 = __builtin_amdgcn_wmma_f32_16x16x32_f16(false, af, false, b2, (short)0, acc2, false, false);
    acc3 = __builtin_amdgcn_wmma_f32_16x16x32_f16(false, af, false, b3, (short)0, acc3, false, false);
  }
#pragma unroll
  for (int t = 0; t < 4; ++t) {
    const v8f acc = (t == 0) ? acc0 : (t == 1) ? acc1 : (t == 2) ? acc2 : acc3;
    const int n = nb + t * 16 + l15;
    const float bn = bias[n];
#pragma unroll
    for (int i = 0; i < 8; ++i) {
      const int m = m0 + half * 8 + i;             // C layout: VGPR i = M i / 8+i
      float v = acc[i] + bn;
      if (MODE == 1) v = fmaxf(v, 0.0f);
      if (MODE >= 2) v += Res[(size_t)m * N + n];
      if (MODE == 3 && logs[m] == 0) v = 0.0f;
      Cout[(size_t)m * N + n] = v;
    }
  }
}

// ---------------------------------------------------------------- attention
// grid (B*H, S/8); one wave per query. O may alias Q (each (b,q,h) slice is
// read into LDS by its owning wave before any O write).
__global__ __launch_bounds__(256)
void attn_kernel(const float* __restrict__ Q, const float* __restrict__ Kb,
                 const float* __restrict__ Vb, float* __restrict__ O) {
  __shared__ float sq[8][DHD];
  __shared__ float sp[8][SS];
  const int lane = threadIdx.x & 31;
  const int wave = threadIdx.x >> 5;
  const int b = blockIdx.x / HH;
  const int h = blockIdx.x % HH;
  const int qi = blockIdx.y * 8 + wave;            // S = 200 = 25*8, always valid
  const size_t rowq = ((size_t)(b * SS + qi)) * DD + h * DHD;
  sq[wave][lane]      = Q[rowq + lane];
  sq[wave][lane + 32] = Q[rowq + lane + 32];
  __syncthreads();
  float sc[7];
  float mx = -3.0e38f;
#pragma unroll
  for (int t = 0; t < 7; ++t) {
    const int k = t * 32 + lane;
    float dot = -3.0e38f;
    if (k < SS) {
      const float* kr = Kb + ((size_t)(b * SS + k)) * DD + h * DHD;
      float acc = 0.0f;
#pragma unroll 8
      for (int d = 0; d < DHD; ++d) acc += sq[wave][d] * kr[d];
      dot = acc * 0.125f;                          // 1/sqrt(64)
      mx = fmaxf(mx, dot);
    }
    sc[t] = dot;
  }
#pragma unroll
  for (int off = 16; off > 0; off >>= 1) mx = fmaxf(mx, __shfl_xor(mx, off, 32));
  float sum = 0.0f;
#pragma unroll
  for (int t = 0; t < 7; ++t) {
    const int k = t * 32 + lane;
    const float e = (k < SS) ? expf(sc[t] - mx) : 0.0f;
    sc[t] = e; sum += e;
  }
#pragma unroll
  for (int off = 16; off > 0; off >>= 1) sum += __shfl_xor(sum, off, 32);
  const float inv = 1.0f / sum;
#pragma unroll
  for (int t = 0; t < 7; ++t) {
    const int k = t * 32 + lane;
    if (k < SS) sp[wave][k] = sc[t] * inv;
  }
  __syncthreads();
  float o0 = 0.0f, o1 = 0.0f;
  for (int k = 0; k < SS; ++k) {
    const float p = sp[wave][k];
    const float* vr = Vb + ((size_t)(b * SS + k)) * DD + h * DHD;
    o0 += p * vr[lane];
    o1 += p * vr[lane + 32];
  }
  O[rowq + lane]      = o0;
  O[rowq + lane + 32] = o1;
}

// ---------------------------------------------------------------- k-means
// one 256-thread block per batch row; centers/sums live in LDS
__global__ __launch_bounds__(256)
void kmeans_kernel(const float* __restrict__ feats, float* __restrict__ out) {
  __shared__ float cent[KKC][DD];
  __shared__ float sums[KKC][DD];
  __shared__ float cnt[KKC];
  __shared__ int   assignv[SS];
  const int tid = threadIdx.x;
  const int b = blockIdx.x;
  const float* X = feats + (size_t)b * SS * DD;
  for (int i = tid; i < KKC * DD; i += 256) cent[i >> 8][i & 255] = X[i];
  __syncthreads();
  const int lane = tid & 31, wave = tid >> 5;
  for (int it = 0; it < KM_ITERS; ++it) {
    // assignment: one wave per point stripe
    for (int p = wave; p < SS; p += 8) {
      const float* xp = X + (size_t)p * DD;
      float d0 = 0, d1 = 0, d2 = 0, d3 = 0;
      for (int d = lane; d < DD; d += 32) {
        const float xv = xp[d];
        float t;
        t = xv - cent[0][d]; d0 += t * t;
        t = xv - cent[1][d]; d1 += t * t;
        t = xv - cent[2][d]; d2 += t * t;
        t = xv - cent[3][d]; d3 += t * t;
      }
#pragma unroll
      for (int off = 16; off > 0; off >>= 1) {
        d0 += __shfl_xor(d0, off, 32);
        d1 += __shfl_xor(d1, off, 32);
        d2 += __shfl_xor(d2, off, 32);
        d3 += __shfl_xor(d3, off, 32);
      }
      if (lane == 0) {
        int a = 0; float best = d0;                // first-min like jnp.argmin
        if (d1 < best) { best = d1; a = 1; }
        if (d2 < best) { best = d2; a = 2; }
        if (d3 < best) { best = d3; a = 3; }
        assignv[p] = a;
      }
    }
    __syncthreads();
    if (tid < KKC) {
      int c = 0;
      for (int p = 0; p < SS; ++p) c += (assignv[p] == tid);
      cnt[tid] = (float)c;
    }
    {
      const int d = tid;                           // each thread owns one dim
      float s0 = 0, s1 = 0, s2 = 0, s3 = 0;
      for (int p = 0; p < SS; ++p) {
        const float xv = X[(size_t)p * DD + d];
        const int a = assignv[p];
        s0 += (a == 0) ? xv : 0.0f;
        s1 += (a == 1) ? xv : 0.0f;
        s2 += (a == 2) ? xv : 0.0f;
        s3 += (a == 3) ? xv : 0.0f;
      }
      sums[0][d] = s0; sums[1][d] = s1; sums[2][d] = s2; sums[3][d] = s3;
    }
    __syncthreads();
    {
      const int d = tid;
#pragma unroll
      for (int kc = 0; kc < KKC; ++kc)
        if (cnt[kc] > 0.0f) cent[kc][d] = sums[kc][d] / fmaxf(cnt[kc], 1.0f);
    }
    __syncthreads();
  }
  for (int i = tid; i < KKC * DD; i += 256)
    out[(size_t)b * KKC * DD + i] = cent[i >> 8][i & 255];
}

// ---------------------------------------------------------------- launch
extern "C" void kernel_launch(void* const* d_in, const int* in_sizes, int n_in,
                              void* d_out, int out_size, void* d_ws, size_t ws_size,
                              hipStream_t stream) {
  (void)in_sizes; (void)n_in; (void)out_size; (void)ws_size;
  const int*   logs     = (const int*)d_in[0];
  const float* item_emb = (const float*)d_in[1];
  const float* pos_emb  = (const float*)d_in[2];
  const float* ln1_g    = (const float*)d_in[3];
  const float* ln1_b    = (const float*)d_in[4];
  const float* in_w     = (const float*)d_in[5];
  const float* in_b     = (const float*)d_in[6];
  const float* out_w    = (const float*)d_in[7];
  const float* out_b    = (const float*)d_in[8];
  const float* ln2_g    = (const float*)d_in[9];
  const float* ln2_b    = (const float*)d_in[10];
  const float* c1_w     = (const float*)d_in[11];
  const float* c1_b     = (const float*)d_in[12];
  const float* c2_w     = (const float*)d_in[13];
  const float* c2_b     = (const float*)d_in[14];
  const float* last_g   = (const float*)d_in[15];
  const float* last_b   = (const float*)d_in[16];
  float* out = (float*)d_out;

  const size_t NBS = (size_t)BB * SS * DD;         // 6,553,600 elems
  char* ws = (char*)d_ws;
  float* seqs = (float*)(ws + NBS * 4 * 0);
  float* Qb   = (float*)(ws + NBS * 4 * 1);
  float* qb   = (float*)(ws + NBS * 4 * 2);        // also attention output
  float* kb   = (float*)(ws + NBS * 4 * 3);        // also FFN hidden
  float* vb   = (float*)(ws + NBS * 4 * 4);
  f16*   hA   = (f16*)(ws + NBS * 4 * 5);
  f16*   hW   = (f16*)(ws + NBS * 4 * 5 + NBS * 2); // 3*D*D halves max

  const int M = BB * SS;                           // 25600
  const dim3 gGrid(M / 128, DD / 64);              // (200, 4)
  const dim3 gBlk(256);
  const int cvtA = (int)((NBS + 255) / 256);       // 25600 blocks
  const int WDD  = DD * DD;                        // 65536
  const int W3   = 3 * WDD;

  embed_kernel<<<cvtA, 256, 0, stream>>>(logs, item_emb, pos_emb, seqs);

  for (int l = 0; l < LLAYERS; ++l) {
    const float* inw_l = in_w + (size_t)l * W3;
    const float* inb_l = in_b + (size_t)l * 3 * DD;
    // Q = LN1(seqs)
    ln_kernel<<<M / 8, 256, 0, stream>>>(seqs, ln1_g + l * DD, ln1_b + l * DD, Qb, M);
    // all three projection weights to f16 once
    cvt_f32_to_f16<<<(W3 + 255) / 256, 256, 0, stream>>>(inw_l, hW, W3);
    // q = Q @ Wq^T + bq
    cvt_f32_to_f16<<<cvtA, 256, 0, stream>>>(Qb, hA, (int)NBS);
    gemm_wmma_f16<0><<<gGrid, gBlk, 0, stream>>>(hA, hW, inb_l, nullptr, nullptr, qb, M, DD, DD);
    // k, v from raw seqs
    cvt_f32_to_f16<<<cvtA, 256, 0, stream>>>(seqs, hA, (int)NBS);
    gemm_wmma_f16<0><<<gGrid, gBlk, 0, stream>>>(hA, hW + (size_t)WDD, inb_l + DD, nullptr, nullptr, kb, M, DD, DD);
    gemm_wmma_f16<0><<<gGrid, gBlk, 0, stream>>>(hA, hW + (size_t)2 * WDD, inb_l + 2 * DD, nullptr, nullptr, vb, M, DD, DD);
    // softmax attention; output aliases qb
    attn_kernel<<<dim3(BB * HH, SS / 8), 256, 0, stream>>>(qb, kb, vb, qb);
    // seqs = Q + attn @ Wo^T + bo
    cvt_f32_to_f16<<<cvtA, 256, 0, stream>>>(qb, hA, (int)NBS);
    cvt_f32_to_f16<<<(WDD + 255) / 256, 256, 0, stream>>>(out_w + (size_t)l * WDD, hW, WDD);
    gemm_wmma_f16<2><<<gGrid, gBlk, 0, stream>>>(hA, hW, out_b + l * DD, Qb, nullptr, seqs, M, DD, DD);
    // seqs = LN2(seqs) (in place; rows are wave-private)
    ln_kernel<<<M / 8, 256, 0, stream>>>(seqs, ln2_g + l * DD, ln2_b + l * DD, seqs, M);
    // h = relu(seqs @ W1^T + b1)
    cvt_f32_to_f16<<<cvtA, 256, 0, stream>>>(seqs, hA, (int)NBS);
    cvt_f32_to_f16<<<(WDD + 255) / 256, 256, 0, stream>>>(c1_w + (size_t)l * WDD, hW, WDD);
    gemm_wmma_f16<1><<<gGrid, gBlk, 0, stream>>>(hA, hW, c1_b + l * DD, nullptr, nullptr, kb, M, DD, DD);
    // seqs = (h @ W2^T + b2 + seqs) * mask
    cvt_f32_to_f16<<<cvtA, 256, 0, stream>>>(kb, hA, (int)NBS);
    cvt_f32_to_f16<<<(WDD + 255) / 256, 256, 0, stream>>>(c2_w + (size_t)l * WDD, hW, WDD);
    gemm_wmma_f16<3><<<gGrid, gBlk, 0, stream>>>(hA, hW, c2_b + l * DD, seqs, logs, seqs, M, DD, DD);
  }
  // log_feats = LN(seqs)
  ln_kernel<<<M / 8, 256, 0, stream>>>(seqs, last_g, last_b, seqs, M);
  // user_eb = kmeans(log_feats) per batch row
  kmeans_kernel<<<BB, 256, 0, stream>>>(seqs, out);
}
